// SlidingWindowAttention_66408784330768
// MI455X (gfx1250) — matrix operand order repaired and checked by reading, not compiled, not run on hardware
//
#include <hip/hip_runtime.h>

#define H 16
#define HD 64
#define WIN 512
#define DM 1024
#define B 2
#define S 2048
#define M_TOT (B * S)          // 4096 rows
#define N_QKV (3 * DM)         // 3072
#define N64_QKV 48             // 3072/64
#define N64_OUT 16             // 1024/64

typedef __attribute__((ext_vector_type(8)))  _Float16 v8h;
typedef __attribute__((ext_vector_type(16))) _Float16 v16h;
typedef __attribute__((ext_vector_type(8)))  float    v8f;

static __device__ __forceinline__ v8h ld8(const _Float16* p) {
    return *(const v8h*)p;
}
static __device__ __forceinline__ v16h cat16(v8h lo, v8h hi) {
    return __builtin_shufflevector(lo, hi, 0,1,2,3,4,5,6,7,8,9,10,11,12,13,14,15);
}
static __device__ __forceinline__ v8f wmma_f16(v16h a, v16h b, v8f c) {
    return __builtin_amdgcn_wmma_f32_16x16x32_f16(false, a, false, b, (short)0, c, false, false);
}

// ---------------- elementwise converts ----------------
__global__ void cvt_f16_kernel(const float* __restrict__ in, _Float16* __restrict__ out, int n) {
    int i = blockIdx.x * blockDim.x + threadIdx.x;
    if (i < n) out[i] = (_Float16)in[i];
}

// in: K x N row-major f32 -> out: N x K row-major f16 (transposed)
__global__ void transpose_f16_kernel(const float* __restrict__ in, _Float16* __restrict__ out,
                                     int K, int N) {
    int i = blockIdx.x * blockDim.x + threadIdx.x;
    if (i < K * N) {
        int k = i / N, n = i % N;
        out[(size_t)n * K + k] = (_Float16)in[i];
    }
}

// ---------------- QKV projection GEMM (16x64 tile per wave, A reused 4x) ----------------
// xh: (M_TOT x DM) f16 row-major. wT: (N_QKV x DM) f16 (transposed weights).
// Writes q (b,h,s,hd) scaled by 1/8, k (b,h,s,hd), vT (b,h,hd,s).
__global__ __launch_bounds__(128) void qkv_gemm_kernel(
    const _Float16* __restrict__ xh, const _Float16* __restrict__ wT,
    const float* __restrict__ bias,
    _Float16* __restrict__ qb, _Float16* __restrict__ kb, _Float16* __restrict__ vT) {
    int lane = threadIdx.x & 31, wave = threadIdx.x >> 5;
    int tile = blockIdx.x * 4 + wave;            // 256*48 = 12288 tiles
    int mtile = tile / N64_QKV, n64 = tile % N64_QKV;
    int m0 = mtile * 16, n0 = n64 * 64;
    int half = lane >> 4;
    int mrow = lane & 15, ncol = lane & 15;

    const _Float16* ap  = xh + (size_t)(m0 + mrow) * DM + 16 * half;
    const _Float16* bp0 = wT + (size_t)(n0 +  0 + ncol) * DM + 16 * half;
    const _Float16* bp1 = wT + (size_t)(n0 + 16 + ncol) * DM + 16 * half;
    const _Float16* bp2 = wT + (size_t)(n0 + 32 + ncol) * DM + 16 * half;
    const _Float16* bp3 = wT + (size_t)(n0 + 48 + ncol) * DM + 16 * half;

    v8f c0 = {}, c1 = {}, c2 = {}, c3 = {};
#pragma unroll 2
    for (int kk = 0; kk < DM; kk += 32) {
        __builtin_prefetch(ap  + kk + 256, 0, 3);
        __builtin_prefetch(bp0 + kk + 256, 0, 3);
        __builtin_prefetch(bp1 + kk + 256, 0, 3);
        __builtin_prefetch(bp2 + kk + 256, 0, 3);
        __builtin_prefetch(bp3 + kk + 256, 0, 3);
        v16h a  = cat16(ld8(ap  + kk), ld8(ap  + kk + 8));
        v16h b0 = cat16(ld8(bp0 + kk), ld8(bp0 + kk + 8));
        v16h b1 = cat16(ld8(bp1 + kk), ld8(bp1 + kk + 8));
        v16h b2 = cat16(ld8(bp2 + kk), ld8(bp2 + kk + 8));
        v16h b3 = cat16(ld8(bp3 + kk), ld8(bp3 + kk + 8));
        c0 = wmma_f16(a, b0, c0);
        c1 = wmma_f16(a, b1, c1);
        c2 = wmma_f16(a, b2, c2);
        c3 = wmma_f16(a, b3, c3);
    }

    int which = n0 >> 10;          // 0=q 1=k 2=v (uniform: n0 is 64-aligned)
    int h = (n0 >> 6) & 15;        // uniform: 64-wide tile == one head's hd block
    int bidx = m0 >> 11;           // batch (tiles never straddle batch: 2048%16==0)
    int s_base = m0 & 2047;

    v8f acc[4] = {c0, c1, c2, c3};
    if (which == 2) {
        // vT[(b,h,d,s)] : rows v are 8 consecutive s -> one 16B store per subtile
        _Float16* vbase = vT + (size_t)(bidx * H + h) * HD * S;
#pragma unroll
        for (int t = 0; t < 4; ++t) {
            int d = t * 16 + ncol;
            float bs = bias[n0 + t * 16 + ncol];
            v8h pk;
#pragma unroll
            for (int v = 0; v < 8; ++v) pk[v] = (_Float16)(acc[t][v] + bs);
            *(v8h*)(vbase + (size_t)d * S + s_base + 8 * half) = pk;
        }
    } else {
        _Float16* dst = (which == 0) ? qb : kb;
        float scale = (which == 0) ? 0.125f : 1.0f;   // fold 1/sqrt(64) into q
        _Float16* dbase = dst + (size_t)(bidx * H + h) * S * HD;
#pragma unroll
        for (int t = 0; t < 4; ++t) {
            int d = t * 16 + ncol;
            float bs = bias[n0 + t * 16 + ncol];
#pragma unroll
            for (int v = 0; v < 8; ++v) {
                int s = s_base + v + 8 * half;
                dbase[(size_t)s * HD + d] = (_Float16)((acc[t][v] + bs) * scale);
            }
        }
    }
}

// ---------------- sliding-window flash attention ----------------
// q,k: (b,h,s,hd) f16 (q pre-scaled), vT: (b,h,hd,s) f16.
// attn out: (b*s, DM) f16, feature = h*64 + d.
__global__ __launch_bounds__(128) void swa_attn_kernel(
    const _Float16* __restrict__ qb, const _Float16* __restrict__ kb,
    const _Float16* __restrict__ vT, _Float16* __restrict__ attn) {
    __shared__ _Float16 lds[4 * 16 * 40];   // per-wave 16x32 P tile, padded stride 40

    int lane = threadIdx.x & 31, wave = threadIdx.x >> 5;
    int w = blockIdx.x * 4 + wave;          // 4096 waves total
    int b = w >> 11, h = (w >> 7) & 15, qt = w & 127;
    int q0 = qt * 16;
    int half = lane >> 4;
    int mrow = lane & 15, ncol = lane & 15;

    const _Float16* qbh = qb + (size_t)(b * H + h) * S * HD;
    const _Float16* kbh = kb + (size_t)(b * H + h) * S * HD;
    const _Float16* vbh = vT + (size_t)(b * H + h) * HD * S;
    _Float16* ldsw = lds + wave * 16 * 40;

    // Q A-fragments (16 x 64 over hd, split into two K=32 fragments)
    const _Float16* qp = qbh + (size_t)(q0 + mrow) * HD + 16 * half;
    v16h qa0 = cat16(ld8(qp),      ld8(qp + 8));
    v16h qa1 = cat16(ld8(qp + 32), ld8(qp + 40));

    v8f o0 = {}, o1 = {}, o2 = {}, o3 = {};
    float rm[8], rl[8];
#pragma unroll
    for (int v = 0; v < 8; ++v) { rm[v] = -1e30f; rl[v] = 0.0f; }

    int lo = q0 - (WIN - 1); if (lo < 0) lo = 0;
    int kstart = lo & ~31;

    for (int kbk = kstart; kbk <= q0 + 15; kbk += 32) {
        // ---- scores: S[16q x 32k] as two 16x16 D fragments ----
        const _Float16* kp0 = kbh + (size_t)(kbk + ncol) * HD + 16 * half;
        v16h bk00 = cat16(ld8(kp0),      ld8(kp0 + 8));
        v16h bk01 = cat16(ld8(kp0 + 32), ld8(kp0 + 40));
        v8f s0 = {};
        s0 = wmma_f16(qa0, bk00, s0);
        s0 = wmma_f16(qa1, bk01, s0);

        const _Float16* kp1 = kbh + (size_t)(kbk + 16 + ncol) * HD + 16 * half;
        v16h bk10 = cat16(ld8(kp1),      ld8(kp1 + 8));
        v16h bk11 = cat16(ld8(kp1 + 32), ld8(kp1 + 40));
        v8f s1 = {};
        s1 = wmma_f16(qa0, bk10, s1);
        s1 = wmma_f16(qa1, bk11, s1);

        // ---- window mask ----
#pragma unroll
        for (int v = 0; v < 8; ++v) {
            int rq = q0 + v + 8 * half;
            int c0 = kbk + ncol, c1 = kbk + 16 + ncol;
            if (c0 > rq || (rq - c0) >= WIN) s0[v] = -1e30f;
            if (c1 > rq || (rq - c1) >= WIN) s1[v] = -1e30f;
        }

        // ---- online softmax (row stats live in VGPR index; reduce over 16 lanes) ----
#pragma unroll
        for (int v = 0; v < 8; ++v) {
            float mv = fmaxf(s0[v], s1[v]);
            mv = fmaxf(mv, __shfl_xor(mv, 1));
            mv = fmaxf(mv, __shfl_xor(mv, 2));
            mv = fmaxf(mv, __shfl_xor(mv, 4));
            mv = fmaxf(mv, __shfl_xor(mv, 8));
            float mnew = fmaxf(rm[v], mv);
            float scale = __expf(rm[v] - mnew);
            float p0 = __expf(s0[v] - mnew);
            float p1 = __expf(s1[v] - mnew);
            s0[v] = p0; s1[v] = p1;
            float r = p0 + p1;
            r += __shfl_xor(r, 1);
            r += __shfl_xor(r, 2);
            r += __shfl_xor(r, 4);
            r += __shfl_xor(r, 8);
            rl[v] = rl[v] * scale + r;
            rm[v] = mnew;
            o0[v] *= scale; o1[v] *= scale; o2[v] *= scale; o3[v] *= scale;
        }

        // ---- P: D layout -> A layout via LDS (same-wave exchange) ----
#pragma unroll
        for (int v = 0; v < 8; ++v) {
            ldsw[(v + 8 * half) * 40 + ncol]      = (_Float16)s0[v];
            ldsw[(v + 8 * half) * 40 + 16 + ncol] = (_Float16)s1[v];
        }
        asm volatile("s_wait_dscnt 0" ::: "memory");
        const _Float16* pp = ldsw + mrow * 40 + 16 * half;
        v16h pa = cat16(ld8(pp), ld8(pp + 8));

        // ---- P @ V : V B-fragments contiguous over keys via vT ----
        const _Float16* vp = vbh + (size_t)ncol * S + kbk + 16 * half;
        v16h bv0 = cat16(ld8(vp),          ld8(vp + 8));
        v16h bv1 = cat16(ld8(vp + 16 * S), ld8(vp + 16 * S + 8));
        v16h bv2 = cat16(ld8(vp + 32 * S), ld8(vp + 32 * S + 8));
        v16h bv3 = cat16(ld8(vp + 48 * S), ld8(vp + 48 * S + 8));
        o0 = wmma_f16(pa, bv0, o0);
        o1 = wmma_f16(pa, bv1, o1);
        o2 = wmma_f16(pa, bv2, o2);
        o3 = wmma_f16(pa, bv3, o3);
    }

    // ---- normalize + store (row = b*S + q0 + v + 8*half, feature = h*64 + t*16 + ncol) ----
    _Float16* ao = attn + ((size_t)(b * S + q0 + 8 * half)) * DM + h * HD + ncol;
#pragma unroll
    for (int v = 0; v < 8; ++v) {
        float inv = 1.0f / rl[v];
        ao[(size_t)v * DM + 0]  = (_Float16)(o0[v] * inv);
        ao[(size_t)v * DM + 16] = (_Float16)(o1[v] * inv);
        ao[(size_t)v * DM + 32] = (_Float16)(o2[v] * inv);
        ao[(size_t)v * DM + 48] = (_Float16)(o3[v] * inv);
    }
}

// ---------------- output projection GEMM (16x64 tile per wave) ----------------
__global__ __launch_bounds__(128) void out_gemm_kernel(
    const _Float16* __restrict__ ah, const _Float16* __restrict__ wT,
    const float* __restrict__ bias, float* __restrict__ out) {
    int lane = threadIdx.x & 31, wave = threadIdx.x >> 5;
    int tile = blockIdx.x * 4 + wave;            // 256*16 = 4096 tiles
    int mtile = tile / N64_OUT, n64 = tile % N64_OUT;
    int m0 = mtile * 16, n0 = n64 * 64;
    int half = lane >> 4;
    int mrow = lane & 15, ncol = lane & 15;

    const _Float16* ap  = ah + (size_t)(m0 + mrow) * DM + 16 * half;
    const _Float16* bp0 = wT + (size_t)(n0 +  0 + ncol) * DM + 16 * half;
    const _Float16* bp1 = wT + (size_t)(n0 + 16 + ncol) * DM + 16 * half;
    const _Float16* bp2 = wT + (size_t)(n0 + 32 + ncol) * DM + 16 * half;
    const _Float16* bp3 = wT + (size_t)(n0 + 48 + ncol) * DM + 16 * half;

    v8f c0 = {}, c1 = {}, c2 = {}, c3 = {};
#pragma unroll 2
    for (int kk = 0; kk < DM; kk += 32) {
        __builtin_prefetch(ap  + kk + 256, 0, 3);
        __builtin_prefetch(bp0 + kk + 256, 0, 3);
        __builtin_prefetch(bp1 + kk + 256, 0, 3);
        __builtin_prefetch(bp2 + kk + 256, 0, 3);
        __builtin_prefetch(bp3 + kk + 256, 0, 3);
        v16h a  = cat16(ld8(ap  + kk), ld8(ap  + kk + 8));
        v16h b0 = cat16(ld8(bp0 + kk), ld8(bp0 + kk + 8));
        v16h b1 = cat16(ld8(bp1 + kk), ld8(bp1 + kk + 8));
        v16h b2 = cat16(ld8(bp2 + kk), ld8(bp2 + kk + 8));
        v16h b3 = cat16(ld8(bp3 + kk), ld8(bp3 + kk + 8));
        c0 = wmma_f16(a, b0, c0);
        c1 = wmma_f16(a, b1, c1);
        c2 = wmma_f16(a, b2, c2);
        c3 = wmma_f16(a, b3, c3);
    }

    v8f acc[4] = {c0, c1, c2, c3};
#pragma unroll
    for (int t = 0; t < 4; ++t) {
        float bs = bias[n0 + t * 16 + ncol];
#pragma unroll
        for (int v = 0; v < 8; ++v) {
            int m = m0 + v + 8 * half;
            out[(size_t)m * DM + n0 + t * 16 + ncol] = acc[t][v] + bs;
        }
    }
}

extern "C" void kernel_launch(void* const* d_in, const int* in_sizes, int n_in,
                              void* d_out, int out_size, void* d_ws, size_t ws_size,
                              hipStream_t stream) {
    (void)in_sizes; (void)n_in; (void)out_size; (void)ws_size;
    const float* x    = (const float*)d_in[0];
    const float* wqkv = (const float*)d_in[1];
    const float* bqkv = (const float*)d_in[2];
    const float* wout = (const float*)d_in[3];
    const float* bout = (const float*)d_in[4];
    float* out = (float*)d_out;

    char* ws = (char*)d_ws;
    _Float16* xh   = (_Float16*)ws;  ws += (size_t)M_TOT * DM * 2;        // 8 MB
    _Float16* wqT  = (_Float16*)ws;  ws += (size_t)N_QKV * DM * 2;        // 6 MB
    _Float16* woT  = (_Float16*)ws;  ws += (size_t)DM * DM * 2;           // 2 MB
    _Float16* qb   = (_Float16*)ws;  ws += (size_t)B * H * S * HD * 2;    // 8 MB
    _Float16* kb   = (_Float16*)ws;  ws += (size_t)B * H * S * HD * 2;    // 8 MB
    _Float16* vT   = (_Float16*)ws;  ws += (size_t)B * H * HD * S * 2;    // 8 MB
    _Float16* attn = (_Float16*)ws;  ws += (size_t)M_TOT * DM * 2;        // 8 MB

    int nx = M_TOT * DM;
    cvt_f16_kernel<<<(nx + 255) / 256, 256, 0, stream>>>(x, xh, nx);

    int nwq = DM * N_QKV;
    transpose_f16_kernel<<<(nwq + 255) / 256, 256, 0, stream>>>(wqkv, wqT, DM, N_QKV);

    int nwo = DM * DM;
    transpose_f16_kernel<<<(nwo + 255) / 256, 256, 0, stream>>>(wout, woT, DM, DM);

    // 256 M-tiles * 48 N64-tiles = 12288 waves, 4 waves/block
    qkv_gemm_kernel<<<(M_TOT / 16) * N64_QKV / 4, 128, 0, stream>>>(xh, wqT, bqkv, qb, kb, vT);

    // 2*16*128 = 4096 waves, 4 waves/block
    swa_attn_kernel<<<1024, 128, 0, stream>>>(qb, kb, vT, attn);

    // 256 M-tiles * 16 N64-tiles = 4096 waves, 4 waves/block
    out_gemm_kernel<<<(M_TOT / 16) * N64_OUT / 4, 128, 0, stream>>>(attn, woT, bout, out);
}
